// GroupSpectralConv1d_55929064129232
// MI455X (gfx1250) — matrix operand order repaired and checked by reading, not compiled, not run on hardware
//
#include <hip/hip_runtime.h>
#include <math.h>

// ---------------------------------------------------------------------------
// GroupSpectralConv1d for MI455X (gfx1250, wave32, WMMA).
//
// Only modes 0..63 of the 8192-pt rfft are used and the output spectrum is
// zero-padded above mode 63, so the op is two skinny GEMMs around a tiny
// complex channel mix:
//   fwd : X[4096 x 128] = x[4096 x 8192] * T[8192 x 128]   (T = cos / -sin)
//   mix : Y[b,e,o,k]    = sum_i X[b,e,i,k] * Weff[e,i,o,k] (complex, tiny)
//   inv : out[4096x8192]= Y[4096 x 128] * Binv[128 x 8192] (scaled cos/-sin)
//
// HBM traffic ~270 MB -> ~12 us at 23.3 TB/s.  FLOPs 17.2 G; done as f16
// split-precision (hi/lo, Ah*Bh + Ah*Bl + Al*Bh) on v_wmma_f32_16x16x32_f16
// => ~3.1M WMMA instrs, well under the memory time, ~1e-6 relative error.
//
// Workspace layout (requires ws_size >= 16 MiB):
//   [ 0MB) Th   fwd twiddle hi  f16 [128 col][8192 t]
//   [ 2MB) Tl   fwd twiddle lo
//   [ 4MB) Bh   inv twiddle hi  f16 [8192 t][128 col]   (x2 for k>0, row j=1 zeroed)
//   [ 6MB) Bl   inv twiddle lo
//   [ 8MB) Xm   fwd DFT result f32 [4096][128]
//   [10MB) Wre  effective weights re f32 [e][i][o][k]
//   [12MB) Wim  effective weights im
//   [14MB) Yh   mixed spectrum hi f16 [4096][128]
//   [15MB) Yl   mixed spectrum lo
// ---------------------------------------------------------------------------

typedef __attribute__((ext_vector_type(16))) _Float16 v16h;
typedef __attribute__((ext_vector_type(8)))  _Float16 v8h;
typedef __attribute__((ext_vector_type(8)))  float    v8f;

union F16Frag { v16h v; v8h h[2]; };

#define WMMA_F16(A, B, C) \
  __builtin_amdgcn_wmma_f32_16x16x32_f16(false, (A), false, (B), (short)0, (C), false, false)

constexpr int   LEN   = 8192;   // sequence length
constexpr int   NSEQ  = 4096;   // 16 batch * 256 channels
constexpr int   NCOL  = 128;    // 64 modes * {re, im}
constexpr int   KC    = 128;    // K chunk staged in LDS (fwd)

// ---------------------------------------------------------------------------
// Twiddle tables, double-precision sincos, exact phase reduction, f16 hi/lo.
// ---------------------------------------------------------------------------
__global__ void gen_tables_kernel(_Float16* __restrict__ Th, _Float16* __restrict__ Tl,
                                  _Float16* __restrict__ Bh, _Float16* __restrict__ Bl) {
  int idx = blockIdx.x * 256 + threadIdx.x;       // 128 * 8192 threads
  int j = idx >> 13;                              // column 0..127
  int t = idx & (LEN - 1);                        // sample 0..8191
  int k = j >> 1;                                 // mode
  int p = (k * t) & (LEN - 1);                    // exact phase index
  double ang = (double)p * (6.283185307179586476925286766559 / (double)LEN);
  double s, c;
  sincos(ang, &s, &c);
  // rfft: Re = sum x*cos, Im = -sum x*sin
  float fv = (float)((j & 1) ? -s : c);
  _Float16 h  = (_Float16)fv;
  Th[(size_t)j * LEN + t] = h;
  Tl[(size_t)j * LEN + t] = (_Float16)(fv - (float)h);
  // irfft (only modes < 64 nonzero):
  //   x[n] = (1/L)[ Re X0 + 2*sum_{k>=1}(Re Xk cos - Im Xk sin) ]
  // 1/L applied in f32 at the output store; Im of mode 0 is ignored.
  float ck = (k == 0) ? 1.0f : 2.0f;
  float iv = fv * ck;
  if (j == 1) iv = 0.0f;
  _Float16 ih = (_Float16)iv;
  Bh[(size_t)t * NCOL + j] = ih;
  Bl[(size_t)t * NCOL + j] = (_Float16)(iv - (float)ih);
}

// ---------------------------------------------------------------------------
// Forward pruned DFT:  Xm[s][j] = sum_t x[s][t] * T[j][t]
// One WG per 16 sequences; wave w owns output columns [16w, 16w+16).
// x staged via LDS with on-the-fly f32 -> f16 hi/lo split.
// WMMA 16-bit A layout (ISA 7.12.2): lane half -> K base 0/8, elems 8..15 at +16.
// B loaded in the same k order as A: lane half -> K base 0/8, elems 8..15 at +16.
// ---------------------------------------------------------------------------
__global__ void __launch_bounds__(256)
fwd_dft_kernel(const float* __restrict__ x, const _Float16* __restrict__ Th,
               const _Float16* __restrict__ Tl, float* __restrict__ Xm) {
  __shared__ _Float16 Ah[16][KC + 8];
  __shared__ _Float16 Al[16][KC + 8];

  const int tid  = threadIdx.x;
  const int lane = tid & 31;
  const int wave = tid >> 5;            // 0..7
  const int seq0 = blockIdx.x * 16;
  const int n0   = wave * 16;           // this wave's column tile
  const int m    = lane & 15;
  const int half = lane >> 4;

  // staging assignment: row r, 8 consecutive samples at cc
  const int r  = tid >> 4;              // 0..15
  const int cc = (tid & 15) * 8;        // 0..120

  v8f acc = {};

  for (int k0 = 0; k0 < LEN; k0 += KC) {
    const float* src = x + (size_t)(seq0 + r) * LEN + k0 + cc;
    float4 a0 = *(const float4*)(src);
    float4 a1 = *(const float4*)(src + 4);
    float va[8] = {a0.x, a0.y, a0.z, a0.w, a1.x, a1.y, a1.z, a1.w};
    v8h hv, lv;
#pragma unroll
    for (int q = 0; q < 8; ++q) {
      _Float16 h = (_Float16)va[q];
      hv[q] = h;
      lv[q] = (_Float16)(va[q] - (float)h);
    }
    *(v8h*)&Ah[r][cc] = hv;
    *(v8h*)&Al[r][cc] = lv;
    __syncthreads();

#pragma unroll
    for (int kb = 0; kb < KC; kb += 32) {
      F16Frag ah, al, bh, bl;
      const int ab = kb + half * 8;
      ah.h[0] = *(const v8h*)&Ah[m][ab];
      ah.h[1] = *(const v8h*)&Ah[m][ab + 16];
      al.h[0] = *(const v8h*)&Al[m][ab];
      al.h[1] = *(const v8h*)&Al[m][ab + 16];
      const size_t boff = (size_t)(n0 + m) * LEN + (size_t)(k0 + kb + half * 8);
      bh.h[0] = *(const v8h*)(Th + boff);
      bh.h[1] = *(const v8h*)(Th + boff + 16);
      bl.h[0] = *(const v8h*)(Tl + boff);
      bl.h[1] = *(const v8h*)(Tl + boff + 16);
      acc = WMMA_F16(ah.v, bh.v, acc);   // hi*hi
      acc = WMMA_F16(ah.v, bl.v, acc);   // hi*lo
      acc = WMMA_F16(al.v, bh.v, acc);   // lo*hi
    }
    __syncthreads();
  }

  // C/D layout: element rr -> row rr + 8*half, col = lane&15
#pragma unroll
  for (int rr = 0; rr < 8; ++rr) {
    int mm = half * 8 + rr;
    Xm[(size_t)(seq0 + mm) * NCOL + (n0 + m)] = acc[rr];
  }
}

// ---------------------------------------------------------------------------
// Effective complex weights: Weff[e,i,o,k] = lam_k*cw[i,o,k] + (1-lam_k)*env
// env[e,i,o,k] = sum_c codes[e,c] * codew[c,i,o,k]
// ---------------------------------------------------------------------------
__global__ void mix_w_kernel(const float* __restrict__ codes, const float* __restrict__ comw,
                             const float* __restrict__ codew, const float* __restrict__ filt,
                             float* __restrict__ Wre, float* __restrict__ Wim) {
  int idx = blockIdx.x * 256 + threadIdx.x;   // ((e*32+i)*32+o)*64+k
  int k = idx & 63;
  int o = (idx >> 6) & 31;
  int i = (idx >> 11) & 31;
  int e = (idx >> 16) & 7;
  float lam = filt[k] * (1.0f / 6.0f) + 0.5f;
  lam = fminf(fmaxf(lam, 0.0f), 1.0f);
  size_t iok = (((size_t)i * 32 + o) * 64 + k);
  float cre = comw[iok * 2 + 0];
  float cim = comw[iok * 2 + 1];
  float ere = 0.0f, eim = 0.0f;
#pragma unroll
  for (int c = 0; c < 16; ++c) {
    float cv = codes[e * 16 + c];
    size_t ci = ((((size_t)c * 32 + i) * 32 + o) * 64 + k) * 2;
    ere += cv * codew[ci + 0];
    eim += cv * codew[ci + 1];
  }
  float wr = lam * cre + (1.0f - lam) * ere;
  float wi = lam * cim + (1.0f - lam) * eim;
  size_t widx = (((size_t)e * 32 + i) * 32 + o) * 64 + k;
  Wre[widx] = wr;
  Wim[widx] = wi;
}

// ---------------------------------------------------------------------------
// Apply mix: Y[b,e,o,k] = sum_i Xc[b,e,i,k] * Weff[e,i,o,k]; split Y to f16 hi/lo
// ---------------------------------------------------------------------------
__global__ void mix_apply_kernel(const float* __restrict__ Xm, const float* __restrict__ Wre,
                                 const float* __restrict__ Wim, _Float16* __restrict__ Yh,
                                 _Float16* __restrict__ Yl) {
  int idx = blockIdx.x * 256 + threadIdx.x;   // ((b*8+e)*32+o)*64+k
  int k = idx & 63;
  int o = (idx >> 6) & 31;
  int e = (idx >> 11) & 7;
  int b = (idx >> 14) & 15;
  float sre = 0.0f, sim = 0.0f;
  const float* xp  = Xm  + (size_t)(b * 256 + e * 32) * NCOL + 2 * k;
  const float* wrp = Wre + (((size_t)e * 32) * 32 + o) * 64 + k;
  const float* wip = Wim + (((size_t)e * 32) * 32 + o) * 64 + k;
#pragma unroll 4
  for (int i = 0; i < 32; ++i) {
    float xr = xp[(size_t)i * NCOL];
    float xi = xp[(size_t)i * NCOL + 1];
    float wr = wrp[(size_t)i * 2048];   // i-stride = 32*64
    float wi = wip[(size_t)i * 2048];
    sre += xr * wr - xi * wi;
    sim += xr * wi + xi * wr;
  }
  size_t row = (size_t)(b * 256 + e * 32 + o) * NCOL;
  _Float16 hr = (_Float16)sre;
  Yh[row + 2 * k] = hr;
  Yl[row + 2 * k] = (_Float16)(sre - (float)hr);
  _Float16 hi = (_Float16)sim;
  Yh[row + 2 * k + 1] = hi;
  Yl[row + 2 * k + 1] = (_Float16)(sim - (float)hi);
}

// ---------------------------------------------------------------------------
// Inverse pruned DFT: out[s][t] = (1/L) * sum_j Y[s][j] * Binv[j][t]
// K = 128 only -> pure register GEMM; Y and Binv stay L2-resident.
// grid = (256 m-tiles, 64); wave w of block y owns samples [(8y+w)*16, +16)
// ---------------------------------------------------------------------------
__global__ void __launch_bounds__(256)
inv_dft_kernel(const _Float16* __restrict__ Yh, const _Float16* __restrict__ Yl,
               const _Float16* __restrict__ Bh, const _Float16* __restrict__ Bl,
               float* __restrict__ out) {
  const int tid  = threadIdx.x;
  const int lane = tid & 31;
  const int wave = tid >> 5;
  const int seq0 = blockIdx.x * 16;
  const int t0   = (blockIdx.y * 8 + wave) * 16;
  const int m    = lane & 15;
  const int half = lane >> 4;

  v8f acc = {};
#pragma unroll
  for (int kb = 0; kb < NCOL; kb += 32) {
    F16Frag ah, al, bh, bl;
    const size_t arow = (size_t)(seq0 + m) * NCOL + kb + half * 8;
    ah.h[0] = *(const v8h*)(Yh + arow);
    ah.h[1] = *(const v8h*)(Yh + arow + 16);
    al.h[0] = *(const v8h*)(Yl + arow);
    al.h[1] = *(const v8h*)(Yl + arow + 16);
    const size_t brow = (size_t)(t0 + m) * NCOL + kb + half * 8;
    bh.h[0] = *(const v8h*)(Bh + brow);
    bh.h[1] = *(const v8h*)(Bh + brow + 16);
    bl.h[0] = *(const v8h*)(Bl + brow);
    bl.h[1] = *(const v8h*)(Bl + brow + 16);
    acc = WMMA_F16(ah.v, bh.v, acc);
    acc = WMMA_F16(ah.v, bl.v, acc);
    acc = WMMA_F16(al.v, bh.v, acc);
  }

#pragma unroll
  for (int rr = 0; rr < 8; ++rr) {
    int mm = half * 8 + rr;
    out[(size_t)(seq0 + mm) * LEN + (t0 + m)] = acc[rr] * (1.0f / (float)LEN);
  }
}

// ---------------------------------------------------------------------------
extern "C" void kernel_launch(void* const* d_in, const int* in_sizes, int n_in,
                              void* d_out, int out_size, void* d_ws, size_t ws_size,
                              hipStream_t stream) {
  (void)in_sizes; (void)n_in; (void)out_size; (void)ws_size;

  const float* x     = (const float*)d_in[0];   // (16, 256, 8192)
  const float* codes = (const float*)d_in[1];   // (8, 16)
  const float* comw  = (const float*)d_in[2];   // (32, 32, 64, 2)
  const float* codew = (const float*)d_in[3];   // (512, 32, 64, 2)
  const float* filt  = (const float*)d_in[4];   // (64,)
  float* out = (float*)d_out;                   // (16, 256, 8192)

  char* w = (char*)d_ws;                        // needs >= 16 MiB
  _Float16* Th  = (_Float16*)(w + ( 0ull << 20));
  _Float16* Tl  = (_Float16*)(w + ( 2ull << 20));
  _Float16* Bh  = (_Float16*)(w + ( 4ull << 20));
  _Float16* Bl  = (_Float16*)(w + ( 6ull << 20));
  float*    Xm  = (float*)   (w + ( 8ull << 20));
  float*    Wre = (float*)   (w + (10ull << 20));
  float*    Wim = (float*)   (w + (12ull << 20));
  _Float16* Yh  = (_Float16*)(w + (14ull << 20));
  _Float16* Yl  = (_Float16*)(w + (15ull << 20));

  gen_tables_kernel<<<dim3(4096), dim3(256), 0, stream>>>(Th, Tl, Bh, Bl);
  mix_w_kernel<<<dim3(2048), dim3(256), 0, stream>>>(codes, comw, codew, filt, Wre, Wim);
  fwd_dft_kernel<<<dim3(256), dim3(256), 0, stream>>>(x, Th, Tl, Xm);
  mix_apply_kernel<<<dim3(1024), dim3(256), 0, stream>>>(Xm, Wre, Wim, Yh, Yl);
  inv_dft_kernel<<<dim3(256, 64), dim3(256), 0, stream>>>(Yh, Yl, Bh, Bl, out);
}
